// HierarchicalAttention_81509889344160
// MI455X (gfx1250) — matrix operand order repaired and checked by reading, not compiled, not run on hardware
//
#include <hip/hip_runtime.h>

typedef __attribute__((ext_vector_type(16))) _Float16 v16h;
typedef __attribute__((ext_vector_type(8)))  float    v8f;
typedef __attribute__((ext_vector_type(4)))  unsigned v4u;
typedef __attribute__((ext_vector_type(4)))  int      v4i;
typedef __attribute__((ext_vector_type(8)))  int      v8i;

#define NB 32768
#define LN_EPS 1e-5f

// ---- Tensor Data Mover availability (guarded: falls back to plain loads) ----
#if __has_builtin(__builtin_amdgcn_tensor_load_to_lds) && __has_builtin(__builtin_amdgcn_s_wait_tensorcnt)
#define GATE_TDM 1
#endif
#if __has_include(<hip/amd_detail/amd_gfx1250_TDM.h>)
#define TDM_SIX_ARGS 1
#endif

// ---------------- WMMA helpers ----------------

static __device__ __forceinline__ v8f wmma_f16(v16h a, v16h b, v8f c) {
  return __builtin_amdgcn_wmma_f32_16x16x32_f16(
      false, a, false, b, (short)0, c, false, false);
}

// A fragment (16x32 f16) from row-major f32 source (ld = leading dim in floats).
// Lane L: M = L&15 ; half h = L>>4 ; per-lane data = two 32B runs -> 4x float4 loads.
static __device__ __forceinline__ v16h load_A_f32(const float* __restrict__ src,
                                                  int ld, int kbase, int lane) {
  int m = lane & 15, h = lane >> 4;
  const float* base = src + (size_t)m * ld + kbase + 8 * h;
  float4 lo0 = *(const float4*)(base + 0);
  float4 lo1 = *(const float4*)(base + 4);
  float4 hi0 = *(const float4*)(base + 16);
  float4 hi1 = *(const float4*)(base + 20);
  v16h a;
  a[0]  = (_Float16)lo0.x; a[1]  = (_Float16)lo0.y; a[2]  = (_Float16)lo0.z; a[3]  = (_Float16)lo0.w;
  a[4]  = (_Float16)lo1.x; a[5]  = (_Float16)lo1.y; a[6]  = (_Float16)lo1.z; a[7]  = (_Float16)lo1.w;
  a[8]  = (_Float16)hi0.x; a[9]  = (_Float16)hi0.y; a[10] = (_Float16)hi0.z; a[11] = (_Float16)hi0.w;
  a[12] = (_Float16)hi1.x; a[13] = (_Float16)hi1.y; a[14] = (_Float16)hi1.z; a[15] = (_Float16)hi1.w;
  return a;
}

// B fragment from pre-packed fragment-major weights: frag = (k/32)*4 + ntile.
// One contiguous 32B per-lane load (2x b128), fully coalesced (1KB per fragment).
static __device__ __forceinline__ v16h load_B_frag(const _Float16* __restrict__ wp,
                                                   int frag, int lane) {
  return *(const v16h*)(wp + ((size_t)frag * 32 + lane) * 16);
}

// ---------------- epilogue helpers (C layout: VGPR r, lane L -> row r+8*(L>>4), col 16*t+(L&15)) ----

static __device__ __forceinline__ float hsum16(float x) {
  x += __shfl_xor(x, 1, 16);
  x += __shfl_xor(x, 2, 16);
  x += __shfl_xor(x, 4, 16);
  x += __shfl_xor(x, 8, 16);
  return x;
}

static __device__ __forceinline__ void add_bias4(v8f acc[4], const float* __restrict__ b, int lane) {
  int c0 = lane & 15;
#pragma unroll
  for (int t = 0; t < 4; ++t) {
    float bb = b[16 * t + c0];
#pragma unroll
    for (int r = 0; r < 8; ++r) acc[t][r] += bb;
  }
}

static __device__ __forceinline__ void layernorm4(v8f acc[4], const float* __restrict__ g,
                                                  const float* __restrict__ b, int lane) {
  int c0 = lane & 15;
#pragma unroll
  for (int r = 0; r < 8; ++r) {
    float s = acc[0][r] + acc[1][r] + acc[2][r] + acc[3][r];
    float q = acc[0][r] * acc[0][r] + acc[1][r] * acc[1][r] +
              acc[2][r] * acc[2][r] + acc[3][r] * acc[3][r];
    s = hsum16(s);
    q = hsum16(q);
    float mean = s * (1.0f / 64.0f);
    float var  = q * (1.0f / 64.0f) - mean * mean;
    float rstd = rsqrtf(var + LN_EPS);
#pragma unroll
    for (int t = 0; t < 4; ++t) {
      int col = 16 * t + c0;
      acc[t][r] = (acc[t][r] - mean) * rstd * g[col] + b[col];
    }
  }
}

static __device__ __forceinline__ void relu4(v8f acc[4]) {
#pragma unroll
  for (int t = 0; t < 4; ++t)
#pragma unroll
    for (int r = 0; r < 8; ++r) acc[t][r] = fmaxf(acc[t][r], 0.0f);
}

static __device__ __forceinline__ void store_tile(float* __restrict__ dst, int ld,
                                                  v8f acc[4], int lane) {
  int c0 = lane & 15, h = lane >> 4;
#pragma unroll
  for (int t = 0; t < 4; ++t)
#pragma unroll
    for (int r = 0; r < 8; ++r)
      dst[(size_t)(r + 8 * h) * ld + 16 * t + c0] = acc[t][r];
}

#ifdef GATE_TDM
// LDS aperture: generic shared pointer low 32 bits == LDS byte offset (ISA 10.2)
static __device__ __forceinline__ unsigned lds_off(const void* p) {
  return (unsigned)(unsigned long long)p;
}

// TDM: DMA a [16 rows x 32 floats] tile (row stride = DIN floats) into LDS.
// D# per CDNA5 ISA ch.8: 2D tile, data_size=4B, type=2, count=1.
template <int DIN>
static __device__ __forceinline__ void tdm_load_tile(const float* gsrc, unsigned lds_byte) {
  unsigned long long ga = (unsigned long long)gsrc;
  v4u g0;
  g0.x = 1u;                                               // count=1, user mode
  g0.y = lds_byte;                                         // lds_addr
  g0.z = (unsigned)ga;                                     // global_addr[31:0]
  g0.w = (unsigned)((ga >> 32) & 0x01FFFFFFu) | (2u << 30); // addr[56:32] | type=2
  v8i g1;
  g1[0] = (int)(2u << 16);                                 // data_size=4B, mask=0
  g1[1] = (int)(((unsigned)DIN & 0xFFFFu) << 16);          // tensor_dim0[15:0]
  g1[2] = (int)(((unsigned)DIN >> 16) | (((unsigned)NB & 0xFFFFu) << 16)); // dim0 hi | dim1 lo
  g1[3] = (int)(((unsigned)NB >> 16) | (32u << 16));       // dim1 hi | tile_dim0=32
  g1[4] = 16;                                              // tile_dim1=16, tile_dim2=0
  g1[5] = DIN;                                             // tensor_dim0_stride
  g1[6] = 0;
  g1[7] = 0;
  v4i gz = {0, 0, 0, 0};
#ifdef TDM_SIX_ARGS
  v8i gz8 = {0, 0, 0, 0, 0, 0, 0, 0};
  __builtin_amdgcn_tensor_load_to_lds(g0, g1, gz, gz, gz8, 0);
#else
  __builtin_amdgcn_tensor_load_to_lds(g0, g1, gz, gz, 0);
#endif
}
#endif

// ---------------- kernels ----------------

// Pack row-major f32 [K,64] weights into fragment-major f16 WMMA B layout.
// dst[((kc*4+t)*32 + lane)*16 + e] = w[(kc*32 + (lane>>4)*8 + e + (e>=8 ? 8 : 0))*64 + t*16 + (lane&15)]
__global__ void pack_w_f16(const float* __restrict__ w, _Float16* __restrict__ dst, int K) {
  int i = blockIdx.x * 256 + threadIdx.x;
  if (i >= K * 64) return;
  int e = i & 15;
  int lane = (i >> 4) & 31;
  int frag = i >> 9;
  int kc = frag >> 2, t = frag & 3;
  int h = lane >> 4, n = lane & 15;
  int k = kc * 32 + h * 8 + e + ((e >= 8) ? 8 : 0);
  dst[i] = (_Float16)w[k * 64 + 16 * t + n];
}

__global__ void fill_ones(float* __restrict__ p, int n) {
  int i = blockIdx.x * 256 + threadIdx.x;
  if (i < n) p[i] = 1.0f;
}

// Fused gate: out = LN2(ReLU(LN1(x@W1+b1)) @ W2 + b2) + (x@Wres + bres)
template <int DIN>
__global__ __launch_bounds__(128) void gate_kernel(
    const float* __restrict__ x,
    const _Float16* __restrict__ w1, const float* __restrict__ b1,
    const float* __restrict__ ln1g, const float* __restrict__ ln1b,
    const _Float16* __restrict__ w2, const float* __restrict__ b2,
    const float* __restrict__ ln2g, const float* __restrict__ ln2b,
    const _Float16* __restrict__ wres, const float* __restrict__ bres,
    float* __restrict__ out) {
  __shared__ float hbuf[4][16][72];
  int wave = threadIdx.x >> 5, lane = threadIdx.x & 31;
  int row0 = (blockIdx.x * 4 + wave) * 16;
  const float* xrow = x + (size_t)row0 * DIN;

  v8f zero = {};
  v8f acc1[4], accR[4];
#pragma unroll
  for (int t = 0; t < 4; ++t) { acc1[t] = zero; accR[t] = zero; }

#ifdef GATE_TDM
  // TDM double-buffered staging of A tiles: tensor_load_to_lds + s_wait_tensorcnt
  __shared__ float xstage[4][2][16][32];
  unsigned l0 = lds_off(&xstage[wave][0][0][0]);
  unsigned l1 = lds_off(&xstage[wave][1][0][0]);
  const int NK = DIN / 32;
  tdm_load_tile<DIN>(xrow, l0);
  for (int kc = 0; kc < NK; ++kc) {
    if (kc + 1 < NK) {
      tdm_load_tile<DIN>(xrow + (size_t)(kc + 1) * 32, (kc & 1) ? l0 : l1);
      __builtin_amdgcn_s_wait_tensorcnt(1);   // oldest (current buffer) complete
    } else {
      __builtin_amdgcn_s_wait_tensorcnt(0);
    }
    const float* src = (kc & 1) ? &xstage[wave][1][0][0] : &xstage[wave][0][0][0];
    v16h a = load_A_f32(src, 32, 0, lane);
#pragma unroll
    for (int t = 0; t < 4; ++t) {
      acc1[t] = wmma_f16(a, load_B_frag(w1,   kc * 4 + t, lane), acc1[t]);
      accR[t] = wmma_f16(a, load_B_frag(wres, kc * 4 + t, lane), accR[t]);
    }
  }
#else
  for (int k = 0; k < DIN; k += 32) {
    if (k + 32 < DIN) __builtin_prefetch(xrow + (lane & 15) * DIN + k + 32, 0, 1);
    v16h a = load_A_f32(xrow, DIN, k, lane);
#pragma unroll
    for (int t = 0; t < 4; ++t) {
      acc1[t] = wmma_f16(a, load_B_frag(w1,   (k >> 5) * 4 + t, lane), acc1[t]);
      accR[t] = wmma_f16(a, load_B_frag(wres, (k >> 5) * 4 + t, lane), accR[t]);
    }
  }
#endif

  add_bias4(acc1, b1, lane);
  layernorm4(acc1, ln1g, ln1b, lane);
  relu4(acc1);
  store_tile(&hbuf[wave][0][0], 72, acc1, lane);
  __syncthreads();

  v8f acc2[4];
#pragma unroll
  for (int t = 0; t < 4; ++t) acc2[t] = zero;
#pragma unroll
  for (int k = 0; k < 64; k += 32) {
    v16h a = load_A_f32(&hbuf[wave][0][0], 72, k, lane);
#pragma unroll
    for (int t = 0; t < 4; ++t)
      acc2[t] = wmma_f16(a, load_B_frag(w2, (k >> 5) * 4 + t, lane), acc2[t]);
  }
  add_bias4(acc2, b2, lane);
  layernorm4(acc2, ln2g, ln2b, lane);
  add_bias4(accR, bres, lane);
#pragma unroll
  for (int t = 0; t < 4; ++t)
#pragma unroll
    for (int r = 0; r < 8; ++r) acc2[t][r] += accR[t][r];
  store_tile(out + (size_t)row0 * 64, 64, acc2, lane);
}

// Fused pairwise attention (seq_len==1 => softmax==1 => attended==v; q/k dead)
__global__ __launch_bounds__(64) void pairwise_kernel(
    const float* __restrict__ x1, const float* __restrict__ x2,
    const _Float16* __restrict__ wv, const float* __restrict__ bv,
    const _Float16* __restrict__ wo, const float* __restrict__ bo,
    const _Float16* __restrict__ wfl, const float* __restrict__ bfl,
    const float* __restrict__ flng, const float* __restrict__ flnb,
    float* __restrict__ fused) {
  __shared__ float vbuf[2][16][72];
  __shared__ float cat[2][16][192];
  int wave = threadIdx.x >> 5, lane = threadIdx.x & 31;
  int row0 = (blockIdx.x * 2 + wave) * 16;
  v8f zero = {};

  v8f accV[4];
#pragma unroll
  for (int t = 0; t < 4; ++t) accV[t] = zero;
#pragma unroll
  for (int k = 0; k < 64; k += 32) {
    v16h a = load_A_f32(x2 + (size_t)row0 * 64, 64, k, lane);
#pragma unroll
    for (int t = 0; t < 4; ++t)
      accV[t] = wmma_f16(a, load_B_frag(wv, (k >> 5) * 4 + t, lane), accV[t]);
  }
  add_bias4(accV, bv, lane);
  store_tile(&vbuf[wave][0][0], 72, accV, lane);
  __syncthreads();

  v8f accO[4];
#pragma unroll
  for (int t = 0; t < 4; ++t) accO[t] = zero;
#pragma unroll
  for (int k = 0; k < 64; k += 32) {
    v16h a = load_A_f32(&vbuf[wave][0][0], 72, k, lane);
#pragma unroll
    for (int t = 0; t < 4; ++t)
      accO[t] = wmma_f16(a, load_B_frag(wo, (k >> 5) * 4 + t, lane), accO[t]);
  }
  add_bias4(accO, bo, lane);
  store_tile(&cat[wave][0][0], 192, accO, lane);

  for (int idx = lane; idx < 16 * 64; idx += 32) {
    int r = idx >> 6, c = idx & 63;
    size_t g = (size_t)(row0 + r) * 64 + c;
    float a = x1[g], b = x2[g];
    cat[wave][r][64 + c]  = a * b;
    cat[wave][r][128 + c] = fabsf(a - b);
  }
  __syncthreads();

  v8f accF[4];
#pragma unroll
  for (int t = 0; t < 4; ++t) accF[t] = zero;
#pragma unroll
  for (int k = 0; k < 192; k += 32) {
    v16h a = load_A_f32(&cat[wave][0][0], 192, k, lane);
#pragma unroll
    for (int t = 0; t < 4; ++t)
      accF[t] = wmma_f16(a, load_B_frag(wfl, (k >> 5) * 4 + t, lane), accF[t]);
  }
  add_bias4(accF, bfl, lane);
  layernorm4(accF, flng, flnb, lane);
  relu4(accF);
  store_tile(fused + (size_t)row0 * 64, 64, accF, lane);
}

// Final fusion + output LN
__global__ __launch_bounds__(64) void final_kernel(
    const float* __restrict__ f1, const float* __restrict__ f2, const float* __restrict__ f3,
    const _Float16* __restrict__ wal1, const float* __restrict__ bal1,
    const float* __restrict__ alng, const float* __restrict__ alnb,
    const float* __restrict__ wal2, const float* __restrict__ bal2,
    const _Float16* __restrict__ wgl, const float* __restrict__ bgl,
    const float* __restrict__ glng, const float* __restrict__ glnb,
    const _Float16* __restrict__ wfl1, const float* __restrict__ bfl1,
    const float* __restrict__ fln1g, const float* __restrict__ fln1b,
    const _Float16* __restrict__ wfl2, const float* __restrict__ bfl2,
    const float* __restrict__ fln2g, const float* __restrict__ fln2b,
    const float* __restrict__ normg, const float* __restrict__ normb,
    float* __restrict__ outF, float* __restrict__ outW) {
  __shared__ float cat[2][16][192];
  __shared__ float tbuf[2][16][72];
  int wave = threadIdx.x >> 5, lane = threadIdx.x & 31;
  int row0 = (blockIdx.x * 2 + wave) * 16;
  int c0 = lane & 15, h = lane >> 4;
  v8f zero = {};

  for (int idx = lane; idx < 16 * 64; idx += 32) {
    int r = idx >> 6, c = idx & 63;
    size_t g = (size_t)(row0 + r) * 64 + c;
    cat[wave][r][c]       = f1[g];
    cat[wave][r][64 + c]  = f2[g];
    cat[wave][r][128 + c] = f3[g];
  }
  __syncthreads();

  v8f accA[4], accG[4];
#pragma unroll
  for (int t = 0; t < 4; ++t) { accA[t] = zero; accG[t] = zero; }
#pragma unroll
  for (int k = 0; k < 192; k += 32) {
    v16h a = load_A_f32(&cat[wave][0][0], 192, k, lane);
#pragma unroll
    for (int t = 0; t < 4; ++t) {
      accA[t] = wmma_f16(a, load_B_frag(wal1, (k >> 5) * 4 + t, lane), accA[t]);
      accG[t] = wmma_f16(a, load_B_frag(wgl,  (k >> 5) * 4 + t, lane), accG[t]);
    }
  }

  add_bias4(accA, bal1, lane);
  layernorm4(accA, alng, alnb, lane);
#pragma unroll
  for (int t = 0; t < 4; ++t)
#pragma unroll
    for (int r = 0; r < 8; ++r) accA[t][r] = tanhf(accA[t][r]);

  float aw[8][3];
#pragma unroll
  for (int r = 0; r < 8; ++r) {
    float L0 = 0.f, L1 = 0.f, L2 = 0.f;
#pragma unroll
    for (int t = 0; t < 4; ++t) {
      float av = accA[t][r];
      int n = 16 * t + c0;
      L0 += av * wal2[n * 3 + 0];
      L1 += av * wal2[n * 3 + 1];
      L2 += av * wal2[n * 3 + 2];
    }
    L0 = hsum16(L0) + bal2[0];
    L1 = hsum16(L1) + bal2[1];
    L2 = hsum16(L2) + bal2[2];
    float mx = fmaxf(L0, fmaxf(L1, L2));
    float e0 = __expf(L0 - mx), e1 = __expf(L1 - mx), e2 = __expf(L2 - mx);
    float inv = 1.0f / (e0 + e1 + e2);
    aw[r][0] = e0 * inv; aw[r][1] = e1 * inv; aw[r][2] = e2 * inv;
  }
  if (c0 < 3) {
#pragma unroll
    for (int r = 0; r < 8; ++r) {
      float v = (c0 == 0) ? aw[r][0] : ((c0 == 1) ? aw[r][1] : aw[r][2]);
      outW[(size_t)(row0 + r + 8 * h) * 3 + c0] = v;
    }
  }

  add_bias4(accG, bgl, lane);
  layernorm4(accG, glng, glnb, lane);
#pragma unroll
  for (int t = 0; t < 4; ++t)
#pragma unroll
    for (int r = 0; r < 8; ++r) accG[t][r] = 1.0f / (1.0f + __expf(-accG[t][r]));

  v8f att[4];
#pragma unroll
  for (int t = 0; t < 4; ++t)
#pragma unroll
    for (int r = 0; r < 8; ++r) {
      int col = 16 * t + c0, rr = r + 8 * h;
      float v1 = cat[wave][rr][col];
      float v2 = cat[wave][rr][64 + col];
      float v3 = cat[wave][rr][128 + col];
      att[t][r] = aw[r][0] * v1 + aw[r][1] * v2 + aw[r][2] * v3;
      tbuf[wave][rr][col] = att[t][r] * accG[t][r];
    }
  __syncthreads();

  v8f accH[4];
#pragma unroll
  for (int t = 0; t < 4; ++t) accH[t] = zero;
#pragma unroll
  for (int k = 0; k < 256; k += 32) {
    v16h a = (k < 64) ? load_A_f32(&tbuf[wave][0][0], 72, k, lane)
                      : load_A_f32(&cat[wave][0][0], 192, k - 64, lane);
#pragma unroll
    for (int t = 0; t < 4; ++t)
      accH[t] = wmma_f16(a, load_B_frag(wfl1, (k >> 5) * 4 + t, lane), accH[t]);
  }
  add_bias4(accH, bfl1, lane);
  layernorm4(accH, fln1g, fln1b, lane);
  relu4(accH);
  __syncthreads();
  store_tile(&tbuf[wave][0][0], 72, accH, lane);
  __syncthreads();

  v8f accF[4];
#pragma unroll
  for (int t = 0; t < 4; ++t) accF[t] = zero;
#pragma unroll
  for (int k = 0; k < 64; k += 32) {
    v16h a = load_A_f32(&tbuf[wave][0][0], 72, k, lane);
#pragma unroll
    for (int t = 0; t < 4; ++t)
      accF[t] = wmma_f16(a, load_B_frag(wfl2, (k >> 5) * 4 + t, lane), accF[t]);
  }
  add_bias4(accF, bfl2, lane);
  layernorm4(accF, fln2g, fln2b, lane);
#pragma unroll
  for (int t = 0; t < 4; ++t)
#pragma unroll
    for (int r = 0; r < 8; ++r) accF[t][r] += att[t][r];
  layernorm4(accF, normg, normb, lane);
  store_tile(outF + (size_t)row0 * 64, 64, accF, lane);
}

// ---------------- host launcher ----------------
// d_in order (insertion-order pytree flatten): 0 mri, 1 snp, 2 clinical;
// gate[i]: l1.w,l1.b,ln1.g,ln1.b,l2.w,l2.b,ln2.g,ln2.b,res.w,res.b -> 3..32
// ms(33..44), mc(45..56), sc(57..68): q.w,q.b,k.w,k.b,v.w,v.b,o.w,o.b,fl.w,fl.b,fln.g,fln.b
// final(69..86), norm(87,88)
extern "C" void kernel_launch(void* const* d_in, const int* in_sizes, int n_in,
                              void* d_out, int out_size, void* d_ws, size_t ws_size,
                              hipStream_t stream) {
  (void)in_sizes; (void)n_in; (void)out_size; (void)ws_size;
  const int B = NB;
  char* ws = (char*)d_ws;
  size_t off = 0;
  auto a16 = [&](size_t elems) {
    _Float16* p = (_Float16*)(ws + off);
    off = (off + elems * sizeof(_Float16) + 255) & ~(size_t)255;
    return p;
  };
  auto a32 = [&](size_t elems) {
    float* p = (float*)(ws + off);
    off = (off + elems * sizeof(float) + 255) & ~(size_t)255;
    return p;
  };

  _Float16* g0w1 = a16(512 * 64);  _Float16* g0wr = a16(512 * 64);  _Float16* g0w2 = a16(64 * 64);
  _Float16* g1w1 = a16(1024 * 64); _Float16* g1wr = a16(1024 * 64); _Float16* g1w2 = a16(64 * 64);
  _Float16* g2w1 = a16(128 * 64);  _Float16* g2wr = a16(128 * 64);  _Float16* g2w2 = a16(64 * 64);
  _Float16* msv = a16(64 * 64); _Float16* mso = a16(64 * 64); _Float16* msf = a16(192 * 64);
  _Float16* mcv = a16(64 * 64); _Float16* mco = a16(64 * 64); _Float16* mcf = a16(192 * 64);
  _Float16* scv = a16(64 * 64); _Float16* sco = a16(64 * 64); _Float16* scf = a16(192 * 64);
  _Float16* wal1 = a16(192 * 64); _Float16* wgl = a16(192 * 64);
  _Float16* wfl1 = a16(256 * 64); _Float16* wfl2 = a16(64 * 64);

  float* bm  = a32((size_t)B * 64);
  float* bs  = a32((size_t)B * 64);
  float* bc  = a32((size_t)B * 64);
  float* bms = a32((size_t)B * 64);
  float* bmc = a32((size_t)B * 64);
  float* bsc = a32((size_t)B * 64);

  auto F = [&](int i) { return (const float*)d_in[i]; };
  auto pack = [&](int i, _Float16* dst, int K) {
    pack_w_f16<<<(K * 64 + 255) / 256, 256, 0, stream>>>(F(i), dst, K);
  };
  pack(3,  g0w1, 512);  pack(11, g0wr, 512);  pack(7,  g0w2, 64);
  pack(13, g1w1, 1024); pack(21, g1wr, 1024); pack(17, g1w2, 64);
  pack(23, g2w1, 128);  pack(31, g2wr, 128);  pack(27, g2w2, 64);
  pack(37, msv, 64); pack(39, mso, 64); pack(41, msf, 192);
  pack(49, mcv, 64); pack(51, mco, 64); pack(53, mcf, 192);
  pack(61, scv, 64); pack(63, sco, 64); pack(65, scf, 192);
  pack(69, wal1, 192); pack(75, wgl, 192);
  pack(79, wfl1, 256); pack(83, wfl2, 64);

  gate_kernel<512><<<B / 64, 128, 0, stream>>>(
      F(0), g0w1, F(4), F(5), F(6), g0w2, F(8), F(9), F(10), g0wr, F(12), bm);
  gate_kernel<1024><<<B / 64, 128, 0, stream>>>(
      F(1), g1w1, F(14), F(15), F(16), g1w2, F(18), F(19), F(20), g1wr, F(22), bs);
  gate_kernel<128><<<B / 64, 128, 0, stream>>>(
      F(2), g2w1, F(24), F(25), F(26), g2w2, F(28), F(29), F(30), g2wr, F(32), bc);

  pairwise_kernel<<<B / 32, 64, 0, stream>>>(
      bm, bs, msv, F(38), mso, F(40), msf, F(42), F(43), F(44), bms);
  pairwise_kernel<<<B / 32, 64, 0, stream>>>(
      bm, bc, mcv, F(50), mco, F(52), mcf, F(54), F(55), F(56), bmc);
  pairwise_kernel<<<B / 32, 64, 0, stream>>>(
      bs, bc, scv, F(62), sco, F(64), scf, F(66), F(67), F(68), bsc);

  float* out = (float*)d_out;
  fill_ones<<<(3 * B + 255) / 256, 256, 0, stream>>>(out + (size_t)B * 64, 3 * B);

  final_kernel<<<B / 32, 64, 0, stream>>>(
      bms, bmc, bsc,
      wal1, F(70), F(71), F(72), F(73), F(74),
      wgl,  F(76), F(77), F(78),
      wfl1, F(80), F(81), F(82),
      wfl2, F(84), F(85), F(86),
      F(87), F(88),
      out, out + (size_t)B * 67);
}
